// MultiHeadAttention_39891656245493
// MI455X (gfx1250) — matrix operand order repaired and checked by reading, not compiled, not run on hardware
//
#include <hip/hip_runtime.h>
#include <hip/hip_bf16.h>

typedef __attribute__((ext_vector_type(16))) _Float16 v16h;
typedef __attribute__((ext_vector_type(8)))  _Float16 v8h;
typedef __attribute__((ext_vector_type(8)))  float    v8f;

#define CH_IN 256
#define NHEAD 4
#define HDIM 64
#define LLEN 1024
#define NB 16

// ---------------------------------------------------------------------------
// WMMA helpers. Tile layouts follow CDNA5 ISA §7.12.2 exactly:
//  A (16x32 f16, MxK): lane m = l%16; halves e0..7 -> k = klo..klo+7,
//                      e8..15 -> k = klo+16..klo+23, klo = (l<16?0:8).
//  B (32x16 f16, KxN): lane n = l%16; halves as A: e0..7 -> k = klo..klo+7, e8..15 -> k = klo+16..klo+23.
//  C/D (16x16 f32):    value (m,n): vgpr = m%8, lane = n + 16*(m/8).
// ---------------------------------------------------------------------------
__device__ __forceinline__ v8f wmma_f16(v16h a, v16h b, v8f c) {
  return __builtin_amdgcn_wmma_f32_16x16x32_f16(false, a, false, b, (short)0, c,
                                                false, false);
}

// A-tile from row-major [rows][ld] f16 at (m0, k0): two 16-byte reads/lane.
__device__ __forceinline__ v16h load_tile_a(const _Float16* base, int ld,
                                            int m0, int k0, int lane) {
  int m   = m0 + (lane & 15);
  int klo = k0 + ((lane & 16) ? 8 : 0);
  const _Float16* p = base + (size_t)m * ld + klo;
  v8h lo = *(const v8h*)(p);
  v8h hi = *(const v8h*)(p + 16);
  return __builtin_shufflevector(lo, hi, 0, 1, 2, 3, 4, 5, 6, 7,
                                 8, 9, 10, 11, 12, 13, 14, 15);
}

// B-tile where B(k,n) = mem[n0+n][k0+k] (row-major [N][K] source):
// two 16-byte reads per lane, same k order as the A-tile.
__device__ __forceinline__ v16h load_tile_b(const _Float16* base, int ld,
                                            int n0, int k0, int lane) {
  const _Float16* p =
      base + (size_t)(n0 + (lane & 15)) * ld + k0 + ((lane & 16) ? 8 : 0);
  v8h lo = *(const v8h*)(p);
  v8h hi = *(const v8h*)(p + 16);
  return __builtin_shufflevector(lo, hi, 0, 1, 2, 3, 4, 5, 6, 7,
                                 8, 9, 10, 11, 12, 13, 14, 15);
}

// ---------------------------------------------------------------------------
// Kernel 1: f32 -> f16 conversion (weights)
// ---------------------------------------------------------------------------
__global__ void f32_to_f16_kernel(const float* __restrict__ src,
                                  _Float16* __restrict__ dst, int n) {
  int i = blockIdx.x * 256 + threadIdx.x;
  if (i < n) dst[i] = (_Float16)src[i];
}

// ---------------------------------------------------------------------------
// Kernel 2: GroupNorm stats. One block per (b, g); 8 channels x 1024 spatial.
// mr[(b*32+g)*2] = mean, [+1] = rstd.
// ---------------------------------------------------------------------------
__global__ void gn_stats_kernel(const float* __restrict__ x,
                                float* __restrict__ mr) {
  __shared__ float ssum[256], ssq[256];
  int b = blockIdx.x, g = blockIdx.y;
  const float* base = x + ((size_t)b * CH_IN + g * 8) * LLEN;
  float s = 0.f, q = 0.f;
  for (int idx = threadIdx.x; idx < 8 * LLEN; idx += 256) {
    float v = base[idx];
    s += v;
    q += v * v;
  }
  ssum[threadIdx.x] = s;
  ssq[threadIdx.x] = q;
  __syncthreads();
  for (int st = 128; st > 0; st >>= 1) {
    if (threadIdx.x < st) {
      ssum[threadIdx.x] += ssum[threadIdx.x + st];
      ssq[threadIdx.x] += ssq[threadIdx.x + st];
    }
    __syncthreads();
  }
  if (threadIdx.x == 0) {
    float mean = ssum[0] * (1.f / 8192.f);
    float var = ssq[0] * (1.f / 8192.f) - mean * mean;
    mr[((size_t)b * 32 + g) * 2] = mean;
    mr[((size_t)b * 32 + g) * 2 + 1] = rsqrtf(var + 1e-5f);
  }
}

// ---------------------------------------------------------------------------
// Kernel 3: normalize + affine + transpose: x[b][c][i] -> hT[b][i][c] (f16),
// LDS 32x32 tile so both global sides are coalesced.
// ---------------------------------------------------------------------------
__global__ void gn_norm_t_kernel(const float* __restrict__ x,
                                 const float* __restrict__ gamma,
                                 const float* __restrict__ beta,
                                 const float* __restrict__ mr,
                                 _Float16* __restrict__ hT) {
  __shared__ float tile[32][33];
  int b = blockIdx.z;
  int c0 = blockIdx.x * 32;
  int i0 = blockIdx.y * 32;
  for (int r = threadIdx.y; r < 32; r += 8) {
    tile[r][threadIdx.x] =
        x[((size_t)b * CH_IN + c0 + r) * LLEN + i0 + threadIdx.x];
  }
  __syncthreads();
  for (int r = threadIdx.y; r < 32; r += 8) {
    int i = i0 + r;
    int c = c0 + threadIdx.x;
    int g = c >> 3;
    float mean = mr[((size_t)b * 32 + g) * 2];
    float rstd = mr[((size_t)b * 32 + g) * 2 + 1];
    float v = (tile[threadIdx.x][r] - mean) * rstd * gamma[c] + beta[c];
    hT[((size_t)b * LLEN + i) * CH_IN + c] = (_Float16)v;
  }
}

// ---------------------------------------------------------------------------
// Kernel 4: QKV projection GEMMs via WMMA.
// grid.x = b*4+h, grid.y = query block of 64. 4 waves, 16 rows each.
// Q,K stored [bh][L][64]; V stored transposed [bh][64][L] so the attention
// kernel's B-operands are contiguous.
// ---------------------------------------------------------------------------
__global__ __launch_bounds__(128) void qkv_kernel(
    const _Float16* __restrict__ hT, const _Float16* __restrict__ wq16,
    const _Float16* __restrict__ wk16, const _Float16* __restrict__ wv16,
    const float* __restrict__ bq, const float* __restrict__ bk,
    const float* __restrict__ bv, _Float16* __restrict__ Q,
    _Float16* __restrict__ K, _Float16* __restrict__ V) {
  int lane = threadIdx.x & 31, wave = threadIdx.x >> 5;
  int bh = blockIdx.x;
  int b = bh >> 2, h = bh & 3;
  int i0 = blockIdx.y * 64 + wave * 16;
  const _Float16* A = hT + (size_t)b * LLEN * CH_IN;
  const _Float16* W[3] = {wq16 + (size_t)h * HDIM * CH_IN,
                          wk16 + (size_t)h * HDIM * CH_IN,
                          wv16 + (size_t)h * HDIM * CH_IN};
  v8f acc[3][4] = {};
#pragma unroll
  for (int kc = 0; kc < 8; ++kc) {
    v16h a = load_tile_a(A, CH_IN, i0, kc * 32, lane);
#pragma unroll
    for (int m = 0; m < 3; ++m)
#pragma unroll
      for (int nt = 0; nt < 4; ++nt) {
        v16h bt = load_tile_b(W[m], CH_IN, nt * 16, kc * 32, lane);
        acc[m][nt] = wmma_f16(a, bt, acc[m][nt]);
      }
  }
  int n15 = lane & 15, moff = (lane & 16) ? 8 : 0;
  size_t bhoff = (size_t)bh * LLEN * HDIM;
#pragma unroll
  for (int nt = 0; nt < 4; ++nt) {
    int d = nt * 16 + n15;
    float bqv = bq[h * HDIM + d];
    float bkv = bk[h * HDIM + d];
    float bvv = bv[h * HDIM + d];
    v8h vpack;
#pragma unroll
    for (int c = 0; c < 8; ++c) {
      int row = i0 + moff + c;
      Q[bhoff + (size_t)row * HDIM + d] = (_Float16)(acc[0][nt][c] + bqv);
      K[bhoff + (size_t)row * HDIM + d] = (_Float16)(acc[1][nt][c] + bkv);
      vpack[c] = (_Float16)(acc[2][nt][c] + bvv);
    }
    // transposed V store: 8 consecutive halves along i -> one 16B store
    *(v8h*)(V + bhoff + (size_t)d * LLEN + i0 + moff) = vpack;
  }
}

// ---------------------------------------------------------------------------
// Kernel 5: fused flash attention per (b, head, 64-query block).
// Each wave owns 16 query rows; sweeps keys in chunks of 32 with online
// softmax. Score tiles never leave registers; P re-striped via 1KB LDS/wave.
// ---------------------------------------------------------------------------
__global__ __launch_bounds__(128) void attn_kernel(
    const _Float16* __restrict__ Q, const _Float16* __restrict__ K,
    const _Float16* __restrict__ V, _Float16* __restrict__ OT) {
  __shared__ __align__(16) _Float16 pbuf[4][16][32];
  int lane = threadIdx.x & 31, wave = threadIdx.x >> 5;
  int bh = blockIdx.x;
  int b = bh >> 2, h = bh & 3;
  int i0 = blockIdx.y * 64 + wave * 16;
  size_t bhoff = (size_t)bh * LLEN * HDIM;
  const _Float16* Qb = Q + bhoff;
  const _Float16* Kb = K + bhoff;
  const _Float16* Vb = V + bhoff;

  v16h aQ0 = load_tile_a(Qb, HDIM, i0, 0, lane);
  v16h aQ1 = load_tile_a(Qb, HDIM, i0, 32, lane);

  v8f o[4] = {};
  float rm[8], rs[8];
#pragma unroll
  for (int c = 0; c < 8; ++c) { rm[c] = -1e30f; rs[c] = 0.f; }

  int n15 = lane & 15, moff = (lane & 16) ? 8 : 0;
  const float scale = 0.125f;  // 1/sqrt(64)

  for (int j0 = 0; j0 < LLEN; j0 += 32) {
    // scores: 16q x 32k, accumulate over d=64 (2 WMMA per 16x16 tile)
    v8f s0 = {}, s1 = {};
    {
      v16h bt = load_tile_b(Kb, HDIM, j0, 0, lane);
      s0 = wmma_f16(aQ0, bt, s0);
      bt = load_tile_b(Kb, HDIM, j0, 32, lane);
      s0 = wmma_f16(aQ1, bt, s0);
      bt = load_tile_b(Kb, HDIM, j0 + 16, 0, lane);
      s1 = wmma_f16(aQ0, bt, s1);
      bt = load_tile_b(Kb, HDIM, j0 + 16, 32, lane);
      s1 = wmma_f16(aQ1, bt, s1);
    }
    // online softmax; row m = c + 8*(lane/16), xor masks 1..8 reduce one row
#pragma unroll
    for (int c = 0; c < 8; ++c) {
      float v0 = s0[c] * scale, v1 = s1[c] * scale;
      float mx = fmaxf(v0, v1);
#pragma unroll
      for (int msk = 1; msk <= 8; msk <<= 1)
        mx = fmaxf(mx, __shfl_xor(mx, msk, 32));
      float mnew = fmaxf(rm[c], mx);
      float corr = __expf(rm[c] - mnew);
      rm[c] = mnew;
      float p0 = __expf(v0 - mnew);
      float p1 = __expf(v1 - mnew);
      float ps = p0 + p1;
#pragma unroll
      for (int msk = 1; msk <= 8; msk <<= 1) ps += __shfl_xor(ps, msk, 32);
      rs[c] = rs[c] * corr + ps;
#pragma unroll
      for (int nt = 0; nt < 4; ++nt) o[nt][c] *= corr;
      s0[c] = p0;
      s1[c] = p1;
    }
    // D-layout f32 -> A-layout f16 through LDS
    __syncthreads();
#pragma unroll
    for (int c = 0; c < 8; ++c) {
      pbuf[wave][moff + c][n15] = (_Float16)s0[c];
      pbuf[wave][moff + c][16 + n15] = (_Float16)s1[c];
    }
    __syncthreads();
    v16h aP = load_tile_a(&pbuf[wave][0][0], 32, 0, 0, lane);
    // P(16x32) @ V(32x64)
#pragma unroll
    for (int nt = 0; nt < 4; ++nt) {
      v16h bt = load_tile_b(Vb, LLEN, nt * 16, j0, lane);
      o[nt] = wmma_f16(aP, bt, o[nt]);
    }
  }

  // finalize: divide by row sum, write concat-head layout OT[b][i][h*64+d]
  _Float16* OTb = OT + (size_t)b * LLEN * CH_IN + h * HDIM;
#pragma unroll
  for (int nt = 0; nt < 4; ++nt)
#pragma unroll
    for (int c = 0; c < 8; ++c) {
      int row = i0 + moff + c;
      OTb[(size_t)row * CH_IN + nt * 16 + n15] = (_Float16)(o[nt][c] / rs[c]);
    }
}

// ---------------------------------------------------------------------------
// Kernel 6: output projection + bias + residual (f32 out).
// grid.x = b*4 + cblock, grid.y = i block of 64. Wave: 16 c-rows x 64 i.
// ---------------------------------------------------------------------------
__global__ __launch_bounds__(128) void proj_kernel(
    const _Float16* __restrict__ wp16, const _Float16* __restrict__ OT,
    const float* __restrict__ bproj, const float* __restrict__ x,
    float* __restrict__ out) {
  int lane = threadIdx.x & 31, wave = threadIdx.x >> 5;
  int b = blockIdx.x >> 2;
  int cblk = blockIdx.x & 3;
  int i0 = blockIdx.y * 64;
  int c0 = cblk * 64 + wave * 16;
  const _Float16* OTb = OT + (size_t)b * LLEN * CH_IN;
  v8f acc[4] = {};
#pragma unroll
  for (int kc = 0; kc < 8; ++kc) {
    v16h a = load_tile_a(wp16, CH_IN, c0, kc * 32, lane);
#pragma unroll
    for (int nt = 0; nt < 4; ++nt) {
      v16h bt = load_tile_b(OTb, CH_IN, i0 + nt * 16, kc * 32, lane);
      acc[nt] = wmma_f16(a, bt, acc[nt]);
    }
  }
  int n15 = lane & 15, moff = (lane & 16) ? 8 : 0;
#pragma unroll
  for (int nt = 0; nt < 4; ++nt) {
    int col = i0 + nt * 16 + n15;
#pragma unroll
    for (int c = 0; c < 8; ++c) {
      int row = c0 + moff + c;
      size_t idx = ((size_t)b * CH_IN + row) * LLEN + col;
      out[idx] = acc[nt][c] + bproj[row] + x[idx];
    }
  }
}

// ---------------------------------------------------------------------------
// Host launcher
// ---------------------------------------------------------------------------
extern "C" void kernel_launch(void* const* d_in, const int* in_sizes, int n_in,
                              void* d_out, int out_size, void* d_ws,
                              size_t ws_size, hipStream_t stream) {
  (void)in_sizes; (void)n_in; (void)out_size; (void)ws_size;
  const float* x     = (const float*)d_in[0];
  const float* gamma = (const float*)d_in[1];
  const float* beta  = (const float*)d_in[2];
  const float* wq    = (const float*)d_in[3];
  const float* bq    = (const float*)d_in[4];
  const float* wk    = (const float*)d_in[5];
  const float* bk    = (const float*)d_in[6];
  const float* wv    = (const float*)d_in[7];
  const float* bv    = (const float*)d_in[8];
  const float* wproj = (const float*)d_in[9];
  const float* bproj = (const float*)d_in[10];
  float* out = (float*)d_out;

  const size_t NW = (size_t)NHEAD * HDIM * CH_IN;  // 65536 per w tensor
  char* ws = (char*)d_ws;
  float* mr       = (float*)ws;                         // 16*32*2 f32
  _Float16* wq16  = (_Float16*)(ws + 4096);
  _Float16* wk16  = wq16 + NW;
  _Float16* wv16  = wk16 + NW;
  _Float16* wp16  = wv16 + NW;                          // 256*256
  _Float16* hT    = wp16 + (size_t)CH_IN * CH_IN;       // [B][L][C]
  _Float16* Q     = hT + (size_t)NB * LLEN * CH_IN;     // [B*H][L][64]
  _Float16* K     = Q + (size_t)NB * NHEAD * LLEN * HDIM;
  _Float16* V     = K + (size_t)NB * NHEAD * LLEN * HDIM;  // [B*H][64][L]
  _Float16* OT    = V + (size_t)NB * NHEAD * LLEN * HDIM;  // [B][L][256]

  f32_to_f16_kernel<<<(int)(NW / 256), 256, 0, stream>>>(wq, wq16, (int)NW);
  f32_to_f16_kernel<<<(int)(NW / 256), 256, 0, stream>>>(wk, wk16, (int)NW);
  f32_to_f16_kernel<<<(int)(NW / 256), 256, 0, stream>>>(wv, wv16, (int)NW);
  f32_to_f16_kernel<<<(CH_IN * CH_IN) / 256, 256, 0, stream>>>(
      wproj, wp16, CH_IN * CH_IN);

  gn_stats_kernel<<<dim3(NB, 32), 256, 0, stream>>>(x, mr);
  gn_norm_t_kernel<<<dim3(CH_IN / 32, LLEN / 32, NB), dim3(32, 8), 0,
                     stream>>>(x, gamma, beta, mr, hT);

  qkv_kernel<<<dim3(NB * NHEAD, LLEN / 64), 128, 0, stream>>>(
      hT, wq16, wk16, wv16, bq, bk, bv, Q, K, V);

  attn_kernel<<<dim3(NB * NHEAD, LLEN / 64), 128, 0, stream>>>(Q, K, V, OT);

  proj_kernel<<<dim3(NB * 4, LLEN / 64), 128, 0, stream>>>(wp16, OT, bproj, x,
                                                           out);
}
